// SO3Activation_36962488549537
// MI455X (gfx1250) — matrix operand order repaired and checked by reading, not compiled, not run on hardware
//
#include <hip/hip_runtime.h>
#include <hip/hip_bf16.h>

// SO3Activation fused kernel for gfx1250 (MI455X), round 3.
//   pass1: g[gp,b] = sum_k D[gp,k] * F[b,k]          (bf16 WMMA, f32 acc)
//   act  : a = ACT_CST * s * qw[y(gp)] * tanh(g/s)   (f32, qw uniform per 16-g half)
//   pass2: out[b,r] += sum_gp Dt[r,gp] * a[gp,b]     (bf16 WMMA, f32 acc in VGPRs)
// Round-3: revert to round-1 structure (<256 VGPRs, no s_set_vgpr_msb churn);
// replace ocml tanhf (divergent branches) with hardware v_tanh_f32 /
// branch-free exp2 identity.

#define DIMR   286
#define DIMP   288          // padded dim (9 * 32)
#define NKK    9            // K blocks of 32 over DIMP
#define NDT    18           // dim tiles of 16 over DIMP
#define CHUNK  32           // grid points per chunk (pass2 K = 32, no padding)
#define NITER  8            // chunks per block -> 256 g per block
#define BGRP   4            // batch groups of 64 (B = 256)
#define GGRP   216          // 55296 / 256
#define BLOCK  128          // 4 wave32

typedef __attribute__((ext_vector_type(16))) __bf16 v16bf;
typedef __attribute__((ext_vector_type(8)))  float  v8f;
typedef __attribute__((ext_vector_type(4)))  int    v4i;

union Frag {
    struct { v4i lo, hi; } q;
    v16bf v;
};

__device__ __forceinline__ v4i lds_ld128(const __bf16* base, int byteOff) {
    return *(const v4i*)((const char*)base + byteOff);
}

// branch-free tanh: hardware transcendental if the builtin exists,
// otherwise tanh(x) = 1 - 2/(exp2(2*log2e*x) + 1)  (correct +-1 limits)
__device__ __forceinline__ float fast_tanh(float x) {
#if __has_builtin(__builtin_amdgcn_tanhf)
    return __builtin_amdgcn_tanhf(x);
#else
    float e = __builtin_amdgcn_exp2f(x * 2.88539008f);
    return 1.0f - 2.0f * __builtin_amdgcn_rcpf(e + 1.0f);
#endif
}

__global__ void so3_zero_kernel(float* __restrict__ out, int n) {
    int i = blockIdx.x * blockDim.x + threadIdx.x;
    if (i < n) out[i] = 0.0f;
}

__global__ __launch_bounds__(BLOCK)
void so3act_fused_kernel(const float* __restrict__ Fg,
                         const float* __restrict__ Dg,
                         const float* __restrict__ qw,
                         float* __restrict__ out)
{
    __shared__ __align__(16) __bf16 sF[64 * DIMP];        // 36 KB: F slice, bf16
    __shared__ __align__(16) __bf16 sDrow[CHUNK * DIMP];  // 18 KB: D tile row-major
    __shared__ __align__(16) __bf16 sDt[DIMP * CHUNK];    // 18 KB: D tile transposed
    __shared__ __align__(16) __bf16 sGact[4 * 16 * CHUNK];// 4 KB : per-wave act tiles

    const int tid  = threadIdx.x;
    const int lane = tid & 31;
    const int wave = tid >> 5;
    const int l16  = lane & 15;
    const int hi   = lane >> 4;          // 0: lanes 0-15, 1: lanes 16-31

    const int bg     = blockIdx.x & (BGRP - 1);
    const int gg     = blockIdx.x / BGRP;
    const int bbase  = bg * 64;
    const int gbase0 = gg * (NITER * CHUNK);

    const float s_norm  = 16.911535f;     // sqrt(286)
    const float inv_s   = 0.059131313f;   // 1/sqrt(286)
    const float act_cst = 1.5927658f;     // normalize2mom(tanh)

    // ---- stage features (64 rows, fp32 -> bf16, pad to 288) ----
    for (int t = tid; t < 64 * 143; t += BLOCK) {
        int b = t / 143;
        int d = (t - b * 143) * 2;
        float2 v = *(const float2*)(Fg + (size_t)(bbase + b) * DIMR + d);
        sF[b * DIMP + d]     = (__bf16)v.x;
        sF[b * DIMP + d + 1] = (__bf16)v.y;
    }
    if (tid < 64) {
        sF[tid * DIMP + 286] = (__bf16)0.0f;
        sF[tid * DIMP + 287] = (__bf16)0.0f;
    }
    __syncthreads();

    // ---- hoist F^T B-operands into registers (constant over the G loop) ----
    v16bf Freg[NKK];
#pragma unroll
    for (int kk = 0; kk < NKK; ++kk) {
        int off = (wave * 16 + l16) * (DIMP * 2) + kk * 64 + hi * 16;
        Frag f;
        f.q.lo = lds_ld128(sF, off);
        f.q.hi = lds_ld128(sF, off + 32);
        Freg[kk] = f.v;
    }

    // ---- fp32 output accumulators: 16 batches x 288 dims per wave ----
    v8f acc[NDT];
#pragma unroll
    for (int t = 0; t < NDT; ++t)
#pragma unroll
        for (int j = 0; j < 8; ++j) acc[t][j] = 0.0f;

    __bf16* gactW = sGact + wave * 16 * CHUNK;

#pragma unroll 1
    for (int it = 0; it < NITER; ++it) {
        const int gbase = gbase0 + it * CHUNK;

        // ---- stage D chunk: 32 rows x 288, fp32 -> bf16, row + transposed ----
        for (int t = tid; t < CHUNK * 144; t += BLOCK) {
            int r = t / 144;
            int d = (t - r * 144) * 2;
            float2 v;
            if (d < 286) v = *(const float2*)(Dg + (size_t)(gbase + r) * DIMR + d);
            else { v.x = 0.0f; v.y = 0.0f; }
            __bf16 b0 = (__bf16)v.x, b1 = (__bf16)v.y;
            sDrow[r * DIMP + d]       = b0;
            sDrow[r * DIMP + d + 1]   = b1;
            sDt[d * CHUNK + r]        = b0;
            sDt[(d + 1) * CHUNK + r]  = b1;
        }
        __syncthreads();

        // prefetch next D chunk into cache while we compute
        if (it + 1 < NITER) {
            const float* nxt = Dg + (size_t)(gbase + CHUNK) * DIMR;
            for (int q = tid; q < 143; q += BLOCK)
                __builtin_prefetch(nxt + q * 64, 0, 0);
        }

        // ---- pass 1 + activation, two 16-g halves ----
#pragma unroll
        for (int h = 0; h < 2; ++h) {
            v8f c;
#pragma unroll
            for (int j = 0; j < 8; ++j) c[j] = 0.0f;
#pragma unroll
            for (int kk = 0; kk < NKK; ++kk) {
                int off = (h * 16 + l16) * (DIMP * 2) + kk * 64 + hi * 16;
                Frag a;
                a.q.lo = lds_ld128(sDrow, off);
                a.q.hi = lds_ld128(sDrow, off + 32);
                c = __builtin_amdgcn_wmma_f32_16x16x32_bf16(
                        false, a.v, false, Freg[kk], (short)0, c, false, false);
            }
            // 48 = 3*16 -> all 16 g in a half share the same beta index y
            const int   y     = ((gbase + h * 16) / 48) % 24;
            const float scale = act_cst * s_norm * qw[y];
#pragma unroll
            for (int j = 0; j < 8; ++j) {
                float v = scale * fast_tanh(c[j] * inv_s);
                int   g = h * 16 + j + 8 * hi;           // local grid point
                gactW[l16 * CHUNK + g] = (__bf16)v;      // gact[b][g], bf16
            }
        }

        // ---- pass 2: out += Dt (A) x gact (B), K = 32 grid points ----
        {
            int boff = l16 * (CHUNK * 2) + hi * 16;
            Frag bm;
            bm.q.lo = lds_ld128(gactW, boff);
            bm.q.hi = lds_ld128(gactW, boff + 32);
            v16bf Bop = bm.v;
#pragma unroll
            for (int t = 0; t < NDT; ++t) {
                int aoff = (t * 16 + l16) * (CHUNK * 2) + hi * 16;
                Frag a;
                a.q.lo = lds_ld128(sDt, aoff);
                a.q.hi = lds_ld128(sDt, aoff + 32);
                acc[t] = __builtin_amdgcn_wmma_f32_16x16x32_bf16(
                            false, a.v, false, Bop, (short)0, acc[t], false, false);
            }
        }
        __syncthreads();
    }

    // ---- accumulate wave-private partial sums into d_out ----
    const int bcol = bbase + wave * 16 + l16;
#pragma unroll
    for (int t = 0; t < NDT; ++t) {
#pragma unroll
        for (int j = 0; j < 8; ++j) {
            int r = t * 16 + j + 8 * hi;
            if (r < DIMR)
                unsafeAtomicAdd(out + (size_t)bcol * DIMR + r, acc[t][j]);
        }
    }
}

extern "C" void kernel_launch(void* const* d_in, const int* in_sizes, int n_in,
                              void* d_out, int out_size, void* d_ws, size_t ws_size,
                              hipStream_t stream) {
    const float* F  = (const float*)d_in[0];   // [256, 286] f32
    const float* Dg = (const float*)d_in[1];   // [48, 24, 48, 286] f32
    const float* qw = (const float*)d_in[2];   // [24] f32
    float* out = (float*)d_out;                // [256, 286] f32

    const int n_out = 256 * DIMR;
    so3_zero_kernel<<<(n_out + 255) / 256, 256, 0, stream>>>(out, n_out);
    so3act_fused_kernel<<<BGRP * GGRP, BLOCK, 0, stream>>>(F, Dg, qw, out);
}
